// LowRankSVDBlock_85529978733260
// MI455X (gfx1250) — matrix operand order repaired and checked by reading, not compiled, not run on hardware
//
#include <hip/hip_runtime.h>
#include <hip/hip_bf16.h>

typedef __bf16 bf16;
typedef __attribute__((ext_vector_type(16))) __bf16 v16bf;
typedef __attribute__((ext_vector_type(8)))  __bf16 v8bf;
typedef __attribute__((ext_vector_type(4)))  __bf16 v4bf;
typedef __attribute__((ext_vector_type(8)))  float  v8f;
typedef __attribute__((ext_vector_type(4)))  unsigned int u32x4;
typedef __attribute__((ext_vector_type(8)))  int i32x8;
typedef __attribute__((ext_vector_type(4)))  int i32x4;

#define D_MODEL 1024
#define NHEAD   16
#define DHEAD   64
#define RQKV    32
#define BATCHN  2
#define SEQ     2048
#define DFF     4096
#define NROWS   (BATCHN*SEQ)

__device__ __forceinline__ v8f wmma_bf(v16bf a, v16bf b, v8f c) {
  return __builtin_amdgcn_wmma_f32_16x16x32_bf16(false, a, false, b, (short)0, c,
                                                 false, false);
}

__device__ __forceinline__ v16bf cat16(v8bf lo, v8bf hi) {
  return __builtin_shufflevector(lo, hi, 0, 1, 2, 3, 4, 5, 6, 7,
                                 8, 9, 10, 11, 12, 13, 14, 15);
}

__device__ __forceinline__ float fexp(float x) {          // branch-free exp
  return __builtin_amdgcn_exp2f(x * 1.4426950408889634f);
}
__device__ __forceinline__ float frcp(float x) { return __builtin_amdgcn_rcpf(x); }

__device__ __forceinline__ float gelu_tanh(float x) {     // branch-free tanh-gelu
  float u = 0.7978845608028654f * (x + 0.044715f * x * x * x);
  float e = __builtin_amdgcn_exp2f(u * 2.8853900817779268f);  // exp(2u)
  float t = 1.0f - 2.0f * frcp(e + 1.0f);
  return 0.5f * x * (1.0f + t);
}

__device__ __forceinline__ unsigned ldsOffset(const void* p) {
  return (unsigned)(size_t)p;   // generic LDS pointer low 32 bits = LDS byte offset
}

// ---------------------------------------------------------------------------
// Tensor Data Mover: async 2D tile load Global -> LDS (D# per ISA 8.3/8.4).
// tileW elems contiguous per row, tileH rows, row stride = strideElems.
// LDS padding: padA-coded DWORDs inserted every padI-coded DWORDs.
// ---------------------------------------------------------------------------
__device__ __forceinline__ void tdm_load_2d(const void* g, unsigned ldsOff,
                                            unsigned tileW, unsigned tileH,
                                            unsigned strideElems,
                                            unsigned padI, unsigned padA) {
  unsigned long long ga = (unsigned long long)(size_t)g;
  u32x4 g0 = {1u,                              // count=1, user descriptor
              ldsOff,                          // lds_addr
              (unsigned)ga,                    // global_addr[31:0]
              ((unsigned)(ga >> 32) & 0x01FFFFFFu) | 0x80000000u};  // type=2
  unsigned d0 = (1u << 16)                     // data_size = 2 bytes (bf16)
              | (1u << 20)                     // pad_enable
              | (padI << 22) | (padA << 25);
  unsigned dim0 = strideElems;                 // tensor_dim0 >= tileW (no OOB)
  unsigned dim1 = 0x100000u;                   // tensor_dim1 large (no OOB)
  i32x8 g1 = {(int)d0,
              (int)(dim0 << 16),                       // tensor_dim0[15:0]
              (int)((dim0 >> 16) | (dim1 << 16)),      // dim0 hi | dim1 lo
              (int)((dim1 >> 16) | (tileW << 16)),     // dim1 hi | tile_dim0
              (int)(tileH & 0xFFFFu),                  // tile_dim1 (tile_dim2=0)
              (int)strideElems,                        // tensor_dim0_stride lo32
              0, 0};
  i32x4 z4 = {0, 0, 0, 0};
#if __clang_major__ >= 23
  i32x8 z8 = {0, 0, 0, 0, 0, 0, 0, 0};
  __builtin_amdgcn_tensor_load_to_lds(g0, g1, z4, z4, z8, 0);
#else
  __builtin_amdgcn_tensor_load_to_lds(g0, g1, z4, z4, 0);
#endif
}

// ---------------------------------------------------------------------------
// f32 -> bf16 cast, 4 elems/thread (b128 in, b64 out)
// ---------------------------------------------------------------------------
__global__ __launch_bounds__(256)
void cast_f32_to_bf16(const float* __restrict__ in, bf16* __restrict__ out, long n4) {
  long i = (long)blockIdx.x * blockDim.x + threadIdx.x;
  long stride = (long)gridDim.x * blockDim.x;
  for (; i < n4; i += stride) {
    float4 xv = *(const float4*)(in + i * 4);
    v4bf ov;
    ov[0] = (bf16)xv.x; ov[1] = (bf16)xv.y; ov[2] = (bf16)xv.z; ov[3] = (bf16)xv.w;
    *(v4bf*)(out + i * 4) = ov;
  }
}

// ---------------------------------------------------------------------------
// LayerNorm over D_MODEL, f32 in -> bf16 out. 256 thr/row, 4 elems/thread.
// ---------------------------------------------------------------------------
__global__ __launch_bounds__(256)
void layernorm_bf16(const float* __restrict__ x, const float* __restrict__ w,
                    const float* __restrict__ b, bf16* __restrict__ out) {
  __shared__ float red[256];
  long off = (long)blockIdx.x * D_MODEL;
  int t = threadIdx.x;
  float4 xv = *(const float4*)(x + off + t * 4);
  red[t] = xv.x + xv.y + xv.z + xv.w;
  __syncthreads();
  for (int st = 128; st > 0; st >>= 1) { if (t < st) red[t] += red[t + st]; __syncthreads(); }
  float mean = red[0] * (1.0f / D_MODEL);
  __syncthreads();
  float dx = xv.x - mean, dy = xv.y - mean, dz = xv.z - mean, dw = xv.w - mean;
  red[t] = dx * dx + dy * dy + dz * dz + dw * dw;
  __syncthreads();
  for (int st = 128; st > 0; st >>= 1) { if (t < st) red[t] += red[t + st]; __syncthreads(); }
  float rstd = rsqrtf(red[0] * (1.0f / D_MODEL) + 1e-5f);
  float4 wv = *(const float4*)(w + t * 4);
  float4 bv = *(const float4*)(b + t * 4);
  v4bf ov;
  ov[0] = (bf16)(dx * rstd * wv.x + bv.x);
  ov[1] = (bf16)(dy * rstd * wv.y + bv.y);
  ov[2] = (bf16)(dz * rstd * wv.z + bv.z);
  ov[3] = (bf16)(dw * rstd * wv.w + bv.w);
  *(v4bf*)(out + off + t * 4) = ov;
}

// ---------------------------------------------------------------------------
// Batched BF16 WMMA GEMM, epilogue selected at compile time (EPI template).
// M%128==0, N%64==0, K%32==0 guaranteed by caller. 8 waves as 4(M)x2(N),
// wave tile 32x32 (2x2 accumulators). A tile staged by TDM (double-buffered,
// padded pitch 40), B tile staged transposed via VGPRs (pitch 40).
// EPI: 0=bf16; 1=+bias,bf16; 2=+bias,gelu,bf16; 3=+bias,+resid(f32),f32
// ---------------------------------------------------------------------------
template <int EPI>
__global__ __launch_bounds__(256)
void gemm_bf16_wmma(const bf16* __restrict__ A, long lda, long aOut, long aIn,
                    const bf16* __restrict__ B, long ldb, long bOut, long bIn,
                    const float* __restrict__ bias, long biasIn,
                    const float* __restrict__ resid,
                    bf16* __restrict__ outb, float* __restrict__ outf,
                    long ldc, long cOut, long cIn, int K, int inner) {
  __shared__ bf16 sA[2][128][40];   // [buf][m][k], TDM-padded pitch
  __shared__ bf16 sBt[2][64][40];   // [buf][n][k], transposed B tile

  int z = blockIdx.z;
  int zo = z / inner, zi = z % inner;
  A += (long)zo * aOut + (long)zi * aIn;
  B += (long)zo * bOut + (long)zi * bIn;
  long coff = (long)zo * cOut + (long)zi * cIn;
  long boff = (long)zi * biasIn;

  int wave = threadIdx.x >> 5;
  int lane = threadIdx.x & 31;
  int wm = wave >> 1, wn = wave & 1;             // 4 x 2 wave grid
  int m0blk = blockIdx.x * 128;
  int n0blk = blockIdx.y * 64;
  int row = lane & 15, lh = lane >> 4;

  int bk = threadIdx.x >> 3;                     // B-tile k row 0..31
  int bn = (threadIdx.x & 7) * 8;                // B-tile n segment

  // prologue: TDM A(k=0), VGPR-load B(k=0)
  if (threadIdx.x < 32)
    tdm_load_2d(A + (long)m0blk * lda, ldsOffset(&sA[0][0][0]),
                32, 128, (unsigned)lda, /*16dw*/3, /*4dw*/3);
  v8bf bv = *(const v8bf*)(B + (long)bk * ldb + n0blk + bn);

  v8f acc[2][2] = {};
  int buf = 0;

  for (int k0 = 0; k0 < K; k0 += 32) {
    #pragma unroll
    for (int j = 0; j < 8; ++j) sBt[buf][bn + j][bk] = bv[j];
    if (threadIdx.x < 32) __builtin_amdgcn_s_wait_tensorcnt(0);
    __syncthreads();                              // tiles[buf] ready everywhere

    int kn = (k0 + 32 < K) ? (k0 + 32) : k0;      // clamped next-k (branch-free)
    if (threadIdx.x < 32)
      tdm_load_2d(A + (long)m0blk * lda + kn, ldsOffset(&sA[buf ^ 1][0][0]),
                  32, 128, (unsigned)lda, 3, 3);
    __builtin_prefetch(B + (long)(kn + 32 + bk) * ldb + n0blk + bn, 0, 1);
    v8bf bvn = *(const v8bf*)(B + (long)(kn + bk) * ldb + n0blk + bn);

    v16bf afrag[2], bfrag[2];
    #pragma unroll
    for (int mt = 0; mt < 2; ++mt) {
      const bf16* ap = &sA[buf][wm * 32 + mt * 16 + row][0];
      v8bf lo = *(const v8bf*)(ap + lh * 8);
      v8bf hi = *(const v8bf*)(ap + 16 + lh * 8);
      afrag[mt] = cat16(lo, hi);
    }
    #pragma unroll
    for (int nt = 0; nt < 2; ++nt) {
      const bf16* bp = &sBt[buf][wn * 32 + nt * 16 + row][lh * 8];
      v8bf lo = *(const v8bf*)bp;
      v8bf hi = *(const v8bf*)(bp + 16);
      bfrag[nt] = cat16(lo, hi);
    }
    #pragma unroll
    for (int mt = 0; mt < 2; ++mt)
      #pragma unroll
      for (int nt = 0; nt < 2; ++nt)
        acc[mt][nt] = wmma_bf(afrag[mt], bfrag[nt], acc[mt][nt]);

    bv = bvn;
    buf ^= 1;
  }

  // epilogue; C row = i + 8*lh, col = lane&15 (coalesced across lanes)
  #pragma unroll
  for (int mt = 0; mt < 2; ++mt)
    #pragma unroll
    for (int nt = 0; nt < 2; ++nt)
      #pragma unroll
      for (int i = 0; i < 8; ++i) {
        int r = m0blk + wm * 32 + mt * 16 + lh * 8 + i;
        int c = n0blk + wn * 32 + nt * 16 + row;
        float v = acc[mt][nt][i];
        if (EPI >= 1) v += bias[boff + c];
        if (EPI == 2) v = gelu_tanh(v);
        long idx = coff + (long)r * ldc + c;
        if (EPI == 3) outf[idx] = v + resid[idx];
        else          outb[idx] = (bf16)v;
      }
}

// ---------------------------------------------------------------------------
// Flash attention. Grid: (SEQ/16, BATCHN*NHEAD), 1 wave per block.
// V tile staged by TDM (pad to 72-elem pitch); K fragments direct b128.
// ---------------------------------------------------------------------------
#define SV_PITCH 72
#define SO_PITCH 72

__global__ __launch_bounds__(32)
void attn_flash_wmma(const bf16* __restrict__ Q, const bf16* __restrict__ K,
                     const bf16* __restrict__ V, const unsigned char* __restrict__ amask,
                     bf16* __restrict__ Y) {
  __shared__ float sS[16][32];
  __shared__ bf16  sP[16][32];
  __shared__ bf16  sV[32][SV_PITCH];
  __shared__ bf16  sO[16][SO_PITCH];
  __shared__ float sm[16], sl[16], ssc[16];
  __shared__ unsigned char smask[32];

  int bh = blockIdx.y;
  int b = bh / NHEAD, hd = bh % NHEAD;
  int m0 = blockIdx.x * 16;
  long base = (long)bh * SEQ * DHEAD;
  const bf16* Qp = Q + base;
  const bf16* Kp = K + base;
  const bf16* Vp = V + base;

  int lane = threadIdx.x;
  int row = lane & 15, lh = lane >> 4;
  int col = lane & 15;

  // Q tile 16x64 as two A fragments, kept in registers.
  v16bf qf[2];
  {
    const bf16* qp = Qp + (long)(m0 + row) * DHEAD;
    #pragma unroll
    for (int e = 0; e < 2; ++e) {
      v8bf lo = *(const v8bf*)(qp + e * 32 + lh * 8);
      v8bf hi = *(const v8bf*)(qp + e * 32 + 16 + lh * 8);
      qf[e] = cat16(lo, hi);
    }
  }

  if (lane < 16) { sm[lane] = -1e30f; sl[lane] = 0.f; }
  v8f o[4] = {};
  __syncthreads();

  for (int t0 = 0; t0 <= m0 + 15; t0 += 32) {
    // async stage V tile [32 x 64] via TDM (overlaps with score compute)
    tdm_load_2d(Vp + (long)t0 * DHEAD, ldsOffset(&sV[0][0]),
                64, 32, 64, /*32dw*/4, /*4dw*/3);
    smask[lane] = amask[(long)b * SEQ + t0 + lane];
    __syncthreads();

    // ---- scores: two 16x16 tiles
    #pragma unroll
    for (int c = 0; c < 2; ++c) {
      int tb = t0 + c * 16;
      v8f s = {};
      #pragma unroll
      for (int e = 0; e < 2; ++e) {
        const bf16* kp = Kp + (long)(tb + col) * DHEAD + e * 32 + lh * 8;
        v8bf lo = *(const v8bf*)kp;
        v8bf hi = *(const v8bf*)(kp + 16);
        s = wmma_bf(qf[e], cat16(lo, hi), s);
      }
      // hoisted mask byte (column-only dependence) -> pure cndmask per element
      int mk = (int)smask[c * 16 + col];
      int thr = tb + col - m0 - lh * 8;      // valid iff i >= thr (causal)
      #pragma unroll
      for (int i = 0; i < 8; ++i) {
        int r = lh * 8 + i;
        float v = s[i] * 0.125f;             // 1/sqrt(64)
        bool valid = (i >= thr) & (mk != 0);
        sS[r][c * 16 + col] = valid ? v : -1e9f;
      }
    }
    __syncthreads();

    // ---- online softmax: lanes 0..15 own one row each (branch-free exp2)
    if (lane < 16) {
      int r = lane;
      float mo = sm[r], mx = mo;
      #pragma unroll 4
      for (int j = 0; j < 32; ++j) mx = fmaxf(mx, sS[r][j]);
      float corr = fexp(mo - mx);
      float sum = 0.f;
      #pragma unroll 4
      for (int j = 0; j < 32; ++j) {
        float sv = sS[r][j];
        float p = (sv <= -1e8f) ? 0.f : fexp(sv - mx);
        sP[r][j] = (bf16)p;
        sum += p;
      }
      sm[r] = mx; sl[r] = sl[r] * corr + sum; ssc[r] = corr;
    }
    __syncthreads();

    // ---- O update: rescale, then P(16x32) @ V(32x64) via 4 WMMAs
    v16bf pf;
    {
      v8bf lo = *(const v8bf*)&sP[row][lh * 16];
      v8bf hi = *(const v8bf*)&sP[row][lh * 16 + 8];
      pf = cat16(lo, hi);
    }
    __builtin_amdgcn_s_wait_tensorcnt(0);   // V tile arrived
    #pragma unroll
    for (int nt = 0; nt < 4; ++nt) {
      #pragma unroll
      for (int i = 0; i < 8; ++i) o[nt][i] *= ssc[lh * 8 + i];
      v16bf vf;
      #pragma unroll
      for (int j = 0; j < 16; ++j) vf[j] = sV[lh * 16 + j][nt * 16 + col];
      o[nt] = wmma_bf(pf, vf, o[nt]);
    }
    __syncthreads();
  }

  // ---- finalize: 1/l via rcp, stage in LDS, coalesced b128 stores
  if (lane < 16) ssc[lane] = frcp(sl[lane]);
  __syncthreads();
  #pragma unroll
  for (int nt = 0; nt < 4; ++nt)
    #pragma unroll
    for (int i = 0; i < 8; ++i) {
      int r = lh * 8 + i;
      sO[r][nt * 16 + col] = (bf16)(o[nt][i] * ssc[r]);
    }
  __syncthreads();
  {
    int r = lane >> 1;
    int c0 = (lane & 1) * 32;
    bf16* yp = Y + ((long)b * SEQ + m0 + r) * D_MODEL + hd * DHEAD + c0;
    #pragma unroll
    for (int j = 0; j < 4; ++j)
      *(v8bf*)(yp + j * 8) = *(const v8bf*)&sO[r][c0 + j * 8];
  }
}

// ---------------------------------------------------------------------------
// Host side
// ---------------------------------------------------------------------------
static inline char* ws_alloc(char*& p, size_t bytes) {
  char* r = p;
  p += (bytes + 255) & ~(size_t)255;
  return r;
}

static void gemm(hipStream_t st, int epi,
                 const bf16* A, long lda, long aOut, long aIn,
                 const bf16* B, long ldb, long bOut, long bIn,
                 const float* bias, long biasIn,
                 const float* resid, bf16* outb, float* outf,
                 long ldc, long cOut, long cIn,
                 int M, int N, int K, int nbatch, int inner) {
  dim3 g(M / 128, N / 64, nbatch);
  switch (epi) {
    case 0: gemm_bf16_wmma<0><<<g, 256, 0, st>>>(A, lda, aOut, aIn, B, ldb, bOut, bIn,
             bias, biasIn, resid, outb, outf, ldc, cOut, cIn, K, inner); break;
    case 1: gemm_bf16_wmma<1><<<g, 256, 0, st>>>(A, lda, aOut, aIn, B, ldb, bOut, bIn,
             bias, biasIn, resid, outb, outf, ldc, cOut, cIn, K, inner); break;
    case 2: gemm_bf16_wmma<2><<<g, 256, 0, st>>>(A, lda, aOut, aIn, B, ldb, bOut, bIn,
             bias, biasIn, resid, outb, outf, ldc, cOut, cIn, K, inner); break;
    default: gemm_bf16_wmma<3><<<g, 256, 0, st>>>(A, lda, aOut, aIn, B, ldb, bOut, bIn,
             bias, biasIn, resid, outb, outf, ldc, cOut, cIn, K, inner); break;
  }
}

extern "C" void kernel_launch(void* const* d_in, const int* in_sizes, int n_in,
                              void* d_out, int out_size, void* d_ws, size_t ws_size,
                              hipStream_t stream) {
  const float* hidden = (const float*)d_in[0];
  const unsigned char* amask = (const unsigned char*)d_in[1];
  const float* ln1w = (const float*)d_in[2];
  const float* ln1b = (const float*)d_in[3];
  const float* qU = (const float*)d_in[4];
  const float* qV = (const float*)d_in[5];
  const float* qb = (const float*)d_in[6];
  const float* kU = (const float*)d_in[7];
  const float* kV = (const float*)d_in[8];
  const float* kb = (const float*)d_in[9];
  const float* vU = (const float*)d_in[10];
  const float* vV = (const float*)d_in[11];
  const float* vb = (const float*)d_in[12];
  const float* outU = (const float*)d_in[13];
  const float* outV = (const float*)d_in[14];
  const float* outb = (const float*)d_in[15];
  const float* ln2w = (const float*)d_in[16];
  const float* ln2b = (const float*)d_in[17];
  const float* fc1U = (const float*)d_in[18];
  const float* fc1V = (const float*)d_in[19];
  const float* fc1b = (const float*)d_in[20];
  const float* fc2U = (const float*)d_in[21];
  const float* fc2V = (const float*)d_in[22];
  const float* fc2b = (const float*)d_in[23];

  const long UQKV = (long)D_MODEL * NHEAD * RQKV;
  const long VQKV = (long)NHEAD * RQKV * DHEAD;
  const long UOUT = (long)D_MODEL * 512;
  const long VOUT = (long)512 * D_MODEL;
  const long F1U = (long)D_MODEL * 512;
  const long F1V = (long)512 * DFF;
  const long F2U = (long)DFF * 512;
  const long F2V = (long)512 * D_MODEL;

  char* p = (char*)d_ws;
  bf16* w_qU = (bf16*)ws_alloc(p, UQKV * 2);
  bf16* w_kU = (bf16*)ws_alloc(p, UQKV * 2);
  bf16* w_vU = (bf16*)ws_alloc(p, UQKV * 2);
  bf16* w_qV = (bf16*)ws_alloc(p, VQKV * 2);
  bf16* w_kV = (bf16*)ws_alloc(p, VQKV * 2);
  bf16* w_vV = (bf16*)ws_alloc(p, VQKV * 2);
  bf16* w_oU = (bf16*)ws_alloc(p, UOUT * 2);
  bf16* w_oV = (bf16*)ws_alloc(p, VOUT * 2);
  bf16* w_1U = (bf16*)ws_alloc(p, F1U * 2);
  bf16* w_1V = (bf16*)ws_alloc(p, F1V * 2);
  bf16* w_2U = (bf16*)ws_alloc(p, F2U * 2);
  bf16* w_2V = (bf16*)ws_alloc(p, F2V * 2);

  bf16* xln = (bf16*)ws_alloc(p, (long)NROWS * D_MODEL * 2);
  bf16* Pq  = (bf16*)ws_alloc(p, (long)NROWS * 512 * 2);
  bf16* Pk  = (bf16*)ws_alloc(p, (long)NROWS * 512 * 2);
  bf16* Pv  = (bf16*)ws_alloc(p, (long)NROWS * 512 * 2);
  long qkvN = (long)BATCHN * NHEAD * SEQ * DHEAD;
  bf16* Qb = (bf16*)ws_alloc(p, qkvN * 2);
  bf16* Kb = (bf16*)ws_alloc(p, qkvN * 2);
  bf16* Vb = (bf16*)ws_alloc(p, qkvN * 2);
  bf16* Yb  = (bf16*)ws_alloc(p, (long)NROWS * D_MODEL * 2);
  bf16* t1  = (bf16*)ws_alloc(p, (long)NROWS * 512 * 2);
  float* hbuf = (float*)ws_alloc(p, (long)NROWS * D_MODEL * 4);
  bf16* zb  = (bf16*)ws_alloc(p, (long)NROWS * D_MODEL * 2);
  bf16* tm1 = (bf16*)ws_alloc(p, (long)NROWS * 512 * 2);
  bf16* abuf = (bf16*)ws_alloc(p, (long)NROWS * DFF * 2);
  bf16* tm2 = (bf16*)ws_alloc(p, (long)NROWS * 512 * 2);

  auto cast = [&](const float* src, bf16* dst, long n) {
    cast_f32_to_bf16<<<dim3(256), dim3(256), 0, stream>>>(src, dst, n / 4);
  };
  cast(qU, w_qU, UQKV);  cast(kU, w_kU, UQKV);  cast(vU, w_vU, UQKV);
  cast(qV, w_qV, VQKV);  cast(kV, w_kV, VQKV);  cast(vV, w_vV, VQKV);
  cast(outU, w_oU, UOUT); cast(outV, w_oV, VOUT);
  cast(fc1U, w_1U, F1U); cast(fc1V, w_1V, F1V);
  cast(fc2U, w_2U, F2U); cast(fc2V, w_2V, F2V);

  layernorm_bf16<<<NROWS, 256, 0, stream>>>(hidden, ln1w, ln1b, xln);

  gemm(stream, 0, xln, D_MODEL, 0, 0, w_qU, 512, 0, 0, nullptr, 0, nullptr,
       Pq, nullptr, 512, 0, 0, NROWS, 512, D_MODEL, 1, 1);
  gemm(stream, 0, xln, D_MODEL, 0, 0, w_kU, 512, 0, 0, nullptr, 0, nullptr,
       Pk, nullptr, 512, 0, 0, NROWS, 512, D_MODEL, 1, 1);
  gemm(stream, 0, xln, D_MODEL, 0, 0, w_vU, 512, 0, 0, nullptr, 0, nullptr,
       Pv, nullptr, 512, 0, 0, NROWS, 512, D_MODEL, 1, 1);

  long aO = (long)SEQ * 512, aI = RQKV;
  long bI = (long)RQKV * DHEAD;
  long cO = (long)NHEAD * SEQ * DHEAD, cI = (long)SEQ * DHEAD;
  gemm(stream, 1, Pq, 512, aO, aI, w_qV, DHEAD, 0, bI, qb, DHEAD, nullptr,
       Qb, nullptr, DHEAD, cO, cI, SEQ, DHEAD, RQKV, BATCHN * NHEAD, NHEAD);
  gemm(stream, 1, Pk, 512, aO, aI, w_kV, DHEAD, 0, bI, kb, DHEAD, nullptr,
       Kb, nullptr, DHEAD, cO, cI, SEQ, DHEAD, RQKV, BATCHN * NHEAD, NHEAD);
  gemm(stream, 1, Pv, 512, aO, aI, w_vV, DHEAD, 0, bI, vb, DHEAD, nullptr,
       Vb, nullptr, DHEAD, cO, cI, SEQ, DHEAD, RQKV, BATCHN * NHEAD, NHEAD);

  attn_flash_wmma<<<dim3(SEQ / 16, BATCHN * NHEAD), 32, 0, stream>>>(
      Qb, Kb, Vb, amask, Yb);

  gemm(stream, 0, Yb, D_MODEL, 0, 0, w_oU, 512, 0, 0, nullptr, 0, nullptr,
       t1, nullptr, 512, 0, 0, NROWS, 512, D_MODEL, 1, 1);
  gemm(stream, 3, t1, 512, 0, 0, w_oV, D_MODEL, 0, 0, outb, 0, hidden,
       nullptr, hbuf, D_MODEL, 0, 0, NROWS, D_MODEL, 512, 1, 1);

  layernorm_bf16<<<NROWS, 256, 0, stream>>>(hbuf, ln2w, ln2b, zb);

  gemm(stream, 0, zb, D_MODEL, 0, 0, w_1U, 512, 0, 0, nullptr, 0, nullptr,
       tm1, nullptr, 512, 0, 0, NROWS, 512, D_MODEL, 1, 1);
  gemm(stream, 2, tm1, 512, 0, 0, w_1V, DFF, 0, 0, fc1b, 0, nullptr,
       abuf, nullptr, DFF, 0, 0, NROWS, DFF, 512, 1, 1);
  gemm(stream, 0, abuf, DFF, 0, 0, w_2U, 512, 0, 0, nullptr, 0, nullptr,
       tm2, nullptr, 512, 0, 0, NROWS, 512, DFF, 1, 1);
  gemm(stream, 3, tm2, 512, 0, 0, w_2V, D_MODEL, 0, 0, fc2b, 0, hbuf,
       nullptr, (float*)d_out, D_MODEL, 0, 0, NROWS, D_MODEL, 512, 1, 1);
}
